// AqtConvGeneralDilated_75892072120648
// MI455X (gfx1250) — 8-bit matrix operand order repaired and checked by reading, compiled, not run on hardware
//
#include <hip/hip_runtime.h>
#include <stdint.h>


typedef __attribute__((ext_vector_type(8))) int v8i;
typedef __attribute__((ext_vector_type(2))) int v2i;

#define NB 32
#define HH 56
#define WW 56
#define CI 256
#define CO 256
#define HP 58
#define WP 58
#define KIT 36

static constexpr size_t SZ_QPAD   = (size_t)NB * HP * WP * CI;
static constexpr size_t OFF_BFRAG = SZ_QPAD;
static constexpr size_t SZ_BFRAG  = (size_t)KIT * 16 * 32 * 8 * 4;
static constexpr size_t OFF_AMAXL = OFF_BFRAG + SZ_BFRAG;
static constexpr size_t OFF_AMAXR = OFF_AMAXL + 128;
static constexpr size_t WS_END    = OFF_AMAXR + 1024;
static constexpr int    N_INIT16  = (int)(WS_END / 16);

__global__ void ws_init_kernel(int4* __restrict__ p, int n) {
    int i = blockIdx.x * blockDim.x + threadIdx.x;
    if (i < n) p[i] = make_int4(0, 0, 0, 0);
}

__global__ void amax_lhs_kernel(const float* __restrict__ lhs,
                                unsigned* __restrict__ amaxl) {
    __shared__ float red[256];
    const int n     = blockIdx.y;
    const int chunk = blockIdx.x;
    const float* base = lhs + (size_t)n * (HH * WW * CI) + (size_t)chunk * 6272;
    float m = 0.0f;
    for (int i = threadIdx.x; i < 6272; i += 256)
        m = fmaxf(m, fabsf(base[i]));
    red[threadIdx.x] = m;
    __syncthreads();
    for (int s = 128; s > 0; s >>= 1) {
        if (threadIdx.x < s)
            red[threadIdx.x] = fmaxf(red[threadIdx.x], red[threadIdx.x + s]);
        __syncthreads();
    }
    if (threadIdx.x == 0)
        atomicMax(&amaxl[n], __float_as_uint(red[0]));
}

__global__ void amax_rhs_kernel(const float* __restrict__ rhs,
                                float* __restrict__ amaxr) {
    __shared__ float red[256];
    const int oc = blockIdx.x;
    float m = 0.0f;
    for (int i = threadIdx.x; i < 2304; i += 256)
        m = fmaxf(m, fabsf(rhs[(size_t)i * CO + oc]));
    red[threadIdx.x] = m;
    __syncthreads();
    for (int s = 128; s > 0; s >>= 1) {
        if (threadIdx.x < s)
            red[threadIdx.x] = fmaxf(red[threadIdx.x], red[threadIdx.x + s]);
        __syncthreads();
    }
    if (threadIdx.x == 0) amaxr[oc] = red[0];
}

__global__ void quant_pad_kernel(const float* __restrict__ lhs,
                                 const unsigned* __restrict__ amaxl,
                                 int* __restrict__ qpad) {
    int idx = blockIdx.x * 256 + threadIdx.x;
    int icg = idx & 63;
    int t   = idx >> 6;
    int x   = t % WW;  t /= WW;
    int y   = t % HH;
    int n   = t / HH;
    float inv = 127.0f / fmaxf(__uint_as_float(amaxl[n]), 1e-6f);
    const float* src = lhs + (size_t)idx * 4;
    int packed = 0;
#pragma unroll
    for (int j = 0; j < 4; ++j) {
        int q = (int)rintf(src[j] * inv);
        q = max(-127, min(127, q));
        packed |= (q & 0xff) << (8 * j);
    }
    qpad[((((n * HP) + (y + 1)) * WP) + (x + 1)) * (CI / 4) + icg] = packed;
}

__global__ void pack_b_kernel(const float* __restrict__ rhs,
                              const float* __restrict__ amaxr,
                              int* __restrict__ bfrag) {
    const int bx     = blockIdx.x;
    const int lane   = threadIdx.x;
    const int kIter  = bx >> 4;
    const int octile = bx & 15;
    const int oc = octile * 16 + (lane & 15);
    const int k0 = (lane >> 4) * 8;
    const float inv = 127.0f / fmaxf(amaxr[oc], 1e-6f);
    int* dst = bfrag + ((size_t)bx * 32 + lane) * 8;
#pragma unroll
    for (int v = 0; v < 8; ++v) {
        int kk = kIter * 64 + (v & 1) * 4 + ((v >> 1) << 4) + k0;
        int packed = 0;
#pragma unroll
        for (int j = 0; j < 4; ++j) {
            int q = (int)rintf(rhs[(size_t)(kk + j) * CO + oc] * inv);
            q = max(-127, min(127, q));
            packed |= (q & 0xff) << (8 * j);
        }
        dst[v] = packed;
    }
}

__global__ void __launch_bounds__(256)
conv_wmma_kernel(const int8_t* __restrict__ qpad,
                 const v8i* __restrict__ bfrag,
                 const unsigned* __restrict__ amaxl,
                 const float* __restrict__ amaxr,
                 float* __restrict__ out) {
    const int lane = threadIdx.x & 31;
    const int wave = threadIdx.x >> 5;
    const int job  = blockIdx.x * 8 + wave;
    const int mpair = job >> 2;
    const int ocg   = job & 3;
    const int m0    = mpair << 5;
    const int half  = lane >> 4;
    const int k0A   = half * 8;

    const int8_t* pbase[2];
#pragma unroll
    for (int tl = 0; tl < 2; ++tl) {
        int mr = m0 + tl * 16 + (lane & 15);
        int x = mr % WW;
        int t = mr / WW;
        int y = t % HH;
        int n = t / HH;
        pbase[tl] = qpad + (size_t)(((n * HP + y) * WP) + x) * CI + k0A;
    }

    v8i zero = {0, 0, 0, 0, 0, 0, 0, 0};
    v8i acc[8];
#pragma unroll
    for (int i = 0; i < 8; ++i) acc[i] = zero;

    for (int ky = 0; ky < 3; ++ky) {
#pragma unroll
        for (int kx = 0; kx < 3; ++kx) {
            const int off = (ky * WP + kx) * CI;
            const int kb  = ((ky * 3 + kx) * 4) * 16;
#pragma unroll
            for (int icb = 0; icb < 4; ++icb) {
                v8i a[2];
#pragma unroll
                for (int tl = 0; tl < 2; ++tl) {
                    const int8_t* p2 = pbase[tl] + off + icb * 64;
                    v2i w0 = *(const v2i*)(p2 + 0);
                    v2i w1 = *(const v2i*)(p2 + 16);
                    v2i w2 = *(const v2i*)(p2 + 32);
                    v2i w3 = *(const v2i*)(p2 + 48);
                    a[tl] = (v8i){w0.x, w0.y, w1.x, w1.y,
                                  w2.x, w2.y, w3.x, w3.y};
                }
#pragma unroll
                for (int ot = 0; ot < 4; ++ot) {
                    int fidx = (kb + icb * 16 + ocg * 4 + ot) * 32 + lane;
                    v8i b = bfrag[fidx];
                    acc[ot]     = __builtin_amdgcn_wmma_i32_16x16x64_iu8(
                        true, a[0], true, b, acc[ot], false, false);
                    acc[4 + ot] = __builtin_amdgcn_wmma_i32_16x16x64_iu8(
                        true, a[1], true, b, acc[4 + ot], false, false);
                }
            }
        }
    }

    const float sl = fmaxf(__uint_as_float(amaxl[m0 / (HH * WW)]), 1e-6f) *
                     (1.0f / 127.0f);
#pragma unroll
    for (int ot = 0; ot < 4; ++ot) {
        int oc = (ocg * 4 + ot) * 16 + (lane & 15);
        float s = sl * fmaxf(amaxr[oc], 1e-6f) * (1.0f / 127.0f);
#pragma unroll
        for (int tl = 0; tl < 2; ++tl) {
#pragma unroll
            for (int r = 0; r < 8; ++r) {
                int mo = m0 + tl * 16 + r + half * 8;
                out[(size_t)mo * CO + oc] = (float)acc[tl * 4 + ot][r] * s;
            }
        }
    }
}

extern "C" void kernel_launch(void* const* d_in, const int* in_sizes, int n_in,
                              void* d_out, int out_size, void* d_ws, size_t ws_size,
                              hipStream_t stream) {
    const float* lhs = (const float*)d_in[0];
    const float* rhs = (const float*)d_in[1];
    float* out = (float*)d_out;

    int8_t*   qpad  = (int8_t*)d_ws;
    int*      bfrag = (int*)((char*)d_ws + OFF_BFRAG);
    unsigned* amaxl = (unsigned*)((char*)d_ws + OFF_AMAXL);
    float*    amaxr = (float*)((char*)d_ws + OFF_AMAXR);

    ws_init_kernel<<<(N_INIT16 + 255) / 256, 256, 0, stream>>>((int4*)d_ws, N_INIT16);
    amax_lhs_kernel<<<dim3(128, NB), 256, 0, stream>>>(lhs, amaxl);
    amax_rhs_kernel<<<CO, 256, 0, stream>>>(rhs, amaxr);
    quant_pad_kernel<<<25088, 256, 0, stream>>>(lhs, amaxl, (int*)qpad);
    pack_b_kernel<<<KIT * 16, 32, 0, stream>>>(rhs, amaxr, bfrag);
    conv_wmma_kernel<<<1568, 256, 0, stream>>>(qpad, (const v8i*)bfrag,
                                               amaxl, amaxr, out);
}
